// MultiHeadAttention_43147241456251
// MI455X (gfx1250) — matrix operand order repaired and checked by reading, not compiled, not run on hardware
//
#include <hip/hip_runtime.h>
#include <hip/hip_bf16.h>

// ---------------------------------------------------------------------------
// MHA forward for MI455X (gfx1250, wave32, WMMA + async global->LDS staging).
// ---------------------------------------------------------------------------

typedef _Float16 v4h  __attribute__((ext_vector_type(4)));
typedef _Float16 v8h  __attribute__((ext_vector_type(8)));
typedef _Float16 v16h __attribute__((ext_vector_type(16)));
typedef float    v8f  __attribute__((ext_vector_type(8)));
typedef int      v4i  __attribute__((ext_vector_type(4)));

#define DIMN     1024
#define HEADS    16
#define HEAD_DIM 64
#define BATCH    4
#define SEQ      2048
#define MTOT     (BATCH * SEQ)   // 8192

// ---- async global->LDS (CDNA5 ASYNCcnt path), guarded so compile never fails
#if defined(__HIP_DEVICE_COMPILE__) &&                                         \
    __has_builtin(__builtin_amdgcn_global_load_async_to_lds_b128) &&           \
    __has_builtin(__builtin_amdgcn_s_wait_asynccnt)
#define USE_ASYNC_LDS 1
#else
#define USE_ASYNC_LDS 0
#endif

#if USE_ASYNC_LDS
// exact parameter types per hipcc diagnostic: v4i in AS(1) / AS(3)
typedef __attribute__((address_space(1))) v4i* as1_v4i_ptr;
typedef __attribute__((address_space(3))) v4i* as3_v4i_ptr;
#endif

__device__ __forceinline__ void async_cp16(const void* g, void* l) {
#if USE_ASYNC_LDS
  __builtin_amdgcn_global_load_async_to_lds_b128(
      (as1_v4i_ptr)(uintptr_t)g,
      (as3_v4i_ptr)(unsigned)(uintptr_t)l,
      0, 0);
#else
  (void)g; (void)l;
#endif
}
__device__ __forceinline__ void async_wait_all() {
#if USE_ASYNC_LDS
  __builtin_amdgcn_s_wait_asynccnt(0);
#endif
}

__device__ __forceinline__ v16h cat16(v8h lo, v8h hi) {
  return __builtin_shufflevector(lo, hi, 0,1,2,3,4,5,6,7,8,9,10,11,12,13,14,15);
}

// A fragment (16x32 f16) per CDNA5 ISA 7.12.2:
//   lane m = lane&15, half h = lane>>4
//   elems 0..7  -> K = h*8 + e ; elems 8..15 -> K = 16 + h*8 + (e-8)
__device__ __forceinline__ v16h frag_a(const _Float16* base, int lane, int ld) {
  const int m = lane & 15, h = lane >> 4;
  v8h lo = *reinterpret_cast<const v8h*>(base + m * ld + h * 8);
  v8h hi = *reinterpret_cast<const v8h*>(base + m * ld + 16 + h * 8);
  return cat16(lo, hi);
}

// B fragment (32x16 f16): lane n = lane&15, h = lane>>4, elems 0..7 -> K = h*8 + e; elems 8..15 -> K = 16 + h*8 + (e-8).
// baseT points at K-transposed storage [n][k] with leading dim ld (halfs).
__device__ __forceinline__ v16h frag_b(const _Float16* baseT, int lane, int ld) {
  const int n = lane & 15, h = lane >> 4;
  v8h lo = *reinterpret_cast<const v8h*>(baseT + n * ld + h * 8);
  v8h hi = *reinterpret_cast<const v8h*>(baseT + n * ld + 16 + h * 8);
  return cat16(lo, hi);
}

// ---------------------------------------------------------------------------
// f32 -> f16 conversion (vectorized x4)
// ---------------------------------------------------------------------------
__global__ __launch_bounds__(256) void cvt_f16_x4(const float4* __restrict__ in,
                                                  _Float16* __restrict__ out, int n4) {
  int i = blockIdx.x * 256 + threadIdx.x;
  if (i < n4) {
    float4 f = in[i];
    v4h h;
    h.x = (_Float16)f.x; h.y = (_Float16)f.y; h.z = (_Float16)f.z; h.w = (_Float16)f.w;
    *reinterpret_cast<v4h*>(out + (size_t)i * 4) = h;
  }
}

// ---------------------------------------------------------------------------
// Pack int32 0/1 mask into 1 bit per element (67 MB -> 2 MB: 16 heads re-read
// it from L2 instead of burning ~1 GB of HBM traffic).
// ---------------------------------------------------------------------------
__global__ __launch_bounds__(256) void pack_mask(const int* __restrict__ mask,
                                                 unsigned* __restrict__ bits, int nwords) {
  int w = blockIdx.x * 256 + threadIdx.x;
  if (w < nwords) {
    const int* p = mask + (size_t)w * 32;
    unsigned v = 0u;
#pragma unroll
    for (int j = 0; j < 32; ++j) v |= (p[j] != 0 ? 1u : 0u) << j;
    bits[w] = v;
  }
}

// ---------------------------------------------------------------------------
// WMMA GEMM: C[M,N] = A[M,K](f16) * B[K,N](f16) + bias(f32)
//   block tile 128x128x32, 256 threads = 8 waves (2x4), each wave a 64x32 tile
//   = 4x2 accumulators of v_wmma_f32_16x16x32_f16.
// MODE 0: f32 row-major output; MODE 1: f16 scatter to [B,H,S,Dh] head layout.
// ---------------------------------------------------------------------------
template <int MODE>
__global__ __launch_bounds__(256) void gemm_wmma(const _Float16* __restrict__ A,
                                                 const _Float16* __restrict__ Bm,
                                                 const float* __restrict__ bias,
                                                 void* __restrict__ outp,
                                                 int M, int N, int K) {
  __shared__ _Float16 As[128 * 40];   // [m][k], padded stride 40 halfs (80B)
  __shared__ _Float16 BsT[128 * 40];  // [n][k] (transposed), padded stride 40

  const int tid  = threadIdx.x;
  const int lane = tid & 31;
  const int wave = tid >> 5;
  const int wm = wave >> 2;           // 0..1  -> 64 rows
  const int wn = wave & 3;            // 0..3  -> 32 cols
  const int m0 = blockIdx.y * 128;
  const int n0 = blockIdx.x * 128;

  v8f acc[4][2] = {};

  for (int k0 = 0; k0 < K; k0 += 32) {
    __syncthreads();
    // ---- stage A tile 128x32 (async global->LDS when available) ----
#pragma unroll
    for (int i = 0; i < 2; ++i) {
      int idx = tid + i * 256;
      int r = idx >> 2, c = (idx & 3) * 8;
      const void* g = &A[(size_t)(m0 + r) * K + k0 + c];
#if USE_ASYNC_LDS
      async_cp16(g, &As[r * 40 + c]);
#else
      *reinterpret_cast<v8h*>(&As[r * 40 + c]) = *reinterpret_cast<const v8h*>(g);
#endif
    }
    // ---- stage B tile 32x128 transposed into BsT[n][k] (VGPR path) ----
#pragma unroll
    for (int i = 0; i < 2; ++i) {
      int idx = tid + i * 256;
      int kk = idx >> 4, nn = (idx & 15) * 8;
      v8h v = *reinterpret_cast<const v8h*>(&Bm[(size_t)(k0 + kk) * N + n0 + nn]);
#pragma unroll
      for (int j = 0; j < 8; ++j) BsT[(nn + j) * 40 + kk] = v[j];
    }
    if (k0 + 32 < K) {
      __builtin_prefetch(&A[(size_t)(m0 + (tid >> 1)) * K + k0 + 32], 0, 0);
      __builtin_prefetch(&Bm[(size_t)(k0 + 32 + (tid >> 4)) * N + n0 + (tid & 15) * 8], 0, 0);
    }
    async_wait_all();
    __syncthreads();

    v16h af[4], bf[2];
#pragma unroll
    for (int mt = 0; mt < 4; ++mt)
      af[mt] = frag_a(&As[(wm * 64 + mt * 16) * 40], lane, 40);
#pragma unroll
    for (int nt = 0; nt < 2; ++nt)
      bf[nt] = frag_b(&BsT[(wn * 32 + nt * 16) * 40], lane, 40);
#pragma unroll
    for (int mt = 0; mt < 4; ++mt)
#pragma unroll
      for (int nt = 0; nt < 2; ++nt)
        acc[mt][nt] = __builtin_amdgcn_wmma_f32_16x16x32_f16(
            false, af[mt], false, bf[nt], (short)0, acc[mt][nt], false, false);
  }

  // ---- epilogue: C tile = lane n (lane&15), VGPR r -> M = (lane>>4)*8 + r
  const int nlo = lane & 15, mh = lane >> 4;
#pragma unroll
  for (int mt = 0; mt < 4; ++mt) {
#pragma unroll
    for (int nt = 0; nt < 2; ++nt) {
#pragma unroll
      for (int r = 0; r < 8; ++r) {
        int gm = m0 + wm * 64 + mt * 16 + mh * 8 + r;
        int gn = n0 + wn * 32 + nt * 16 + nlo;
        float val = acc[mt][nt][r] + bias[gn];
        if (MODE == 0) {
          reinterpret_cast<float*>(outp)[(size_t)gm * N + gn] = val;
        } else {
          int b = gm >> 11, s = gm & (SEQ - 1);
          int hh = gn >> 6, d = gn & (HEAD_DIM - 1);
          reinterpret_cast<_Float16*>(outp)[(((size_t)(b * HEADS + hh)) * SEQ + s) * HEAD_DIM + d] =
              (_Float16)val;
        }
      }
    }
  }
}

// ---------------------------------------------------------------------------
// Flash-style attention. 1 block per (b, h, 128 q-rows); 8 waves, each wave
// owns 16 q rows as resident A-fragments. K/V stream through LDS in 64-wide
// blocks (K via async copy); online softmax via shfl_xor over 16-lane halves.
// ---------------------------------------------------------------------------
__global__ __launch_bounds__(256) void attn_wmma(const _Float16* __restrict__ qh,
                                                 const _Float16* __restrict__ kh,
                                                 const _Float16* __restrict__ vh,
                                                 const unsigned* __restrict__ mbits,
                                                 _Float16* __restrict__ ctx) {
  __shared__ _Float16 Kt[64 * 72];       // [kpos][d]  (B-frag ready for Q*K^T)
  __shared__ _Float16 VtT[64 * 72];      // [d][kpos]  (B-frag ready for P*V)
  __shared__ _Float16 Pl[8 * 16 * 72];   // per-wave P staging [16][64]

  const int tid = threadIdx.x, lane = tid & 31, wave = tid >> 5;
  const int blk = blockIdx.x;
  const int qb = (blk & 15) * 128;           // S/128 = 16 q-blocks
  const int h  = (blk >> 4) & (HEADS - 1);
  const int b  = blk >> 8;
  const size_t bh = ((size_t)b * HEADS + h) * SEQ;
  const int qbase = qb + wave * 16;
  const int nlo = lane & 15, mh = lane >> 4;

  // resident Q fragments: 16 rows x 64 dims = 2 A-frags (K-steps of 32)
  v16h aq[2];
  {
    const _Float16* qrow = qh + (bh + qbase + (lane & 15)) * HEAD_DIM;
    const int hl = lane >> 4;
#pragma unroll
    for (int kk = 0; kk < 2; ++kk) {
      v8h lo = *reinterpret_cast<const v8h*>(qrow + kk * 32 + hl * 8);
      v8h hi = *reinterpret_cast<const v8h*>(qrow + kk * 32 + 16 + hl * 8);
      aq[kk] = cat16(lo, hi);
    }
  }

  float m_i[8], l_i[8];
  v8f acc_o[4] = {};
#pragma unroll
  for (int r = 0; r < 8; ++r) { m_i[r] = -1e30f; l_i[r] = 0.0f; }

  for (int kb = 0; kb < SEQ; kb += 64) {
    __syncthreads();
    // ---- stage K (async straight copy) and V (transposed, VGPR path)
#pragma unroll
    for (int i = 0; i < 2; ++i) {
      int idx = tid + i * 256;
      int kp = idx >> 3, dg = (idx & 7) * 8;
      const void* gk = &kh[(bh + kb + kp) * HEAD_DIM + dg];
#if USE_ASYNC_LDS
      async_cp16(gk, &Kt[kp * 72 + dg]);
#else
      *reinterpret_cast<v8h*>(&Kt[kp * 72 + dg]) = *reinterpret_cast<const v8h*>(gk);
#endif
      v8h vv = *reinterpret_cast<const v8h*>(&vh[(bh + kb + kp) * HEAD_DIM + dg]);
#pragma unroll
      for (int j = 0; j < 8; ++j) VtT[(dg + j) * 72 + kp] = vv[j];
    }
    async_wait_all();
    __syncthreads();

    // ---- scores: S(16x64) = Q(16x64) * K^T, 4 tiles x 2 k-steps
    v8f sc[4] = {};
#pragma unroll
    for (int t = 0; t < 4; ++t) {
#pragma unroll
      for (int kk = 0; kk < 2; ++kk) {
        v16h bk = frag_b(&Kt[(t * 16) * 72 + kk * 32], lane, 72);
        sc[t] = __builtin_amdgcn_wmma_f32_16x16x32_f16(
            false, aq[kk], false, bk, (short)0, sc[t], false, false);
      }
    }

    // ---- scale + mask + online softmax (each lane: 8 rows x 1 col x 4 tiles)
    float pv[4][8];
#pragma unroll
    for (int r = 0; r < 8; ++r) {
      int qrow = qbase + mh * 8 + r;
      const unsigned* mr = mbits + ((size_t)b * SEQ + qrow) * (SEQ / 32);
#pragma unroll
      for (int t = 0; t < 4; ++t) {
        int kidx = kb + t * 16 + nlo;
        float sv = sc[t][r] * 0.125f;   // 1/sqrt(HEAD_DIM)
        unsigned w = mr[kidx >> 5];
        if (((w >> (kidx & 31)) & 1u) == 0u) sv = -1e9f;
        pv[t][r] = sv;
      }
      float tm = fmaxf(fmaxf(pv[0][r], pv[1][r]), fmaxf(pv[2][r], pv[3][r]));
#pragma unroll
      for (int off = 8; off >= 1; off >>= 1)
        tm = fmaxf(tm, __shfl_xor(tm, off, 32));
      float nm = fmaxf(m_i[r], tm);
      float corr = __expf(m_i[r] - nm);
      m_i[r] = nm;
      float ps = 0.0f;
#pragma unroll
      for (int t = 0; t < 4; ++t) {
        float p = __expf(pv[t][r] - nm);
        pv[t][r] = p;
        ps += p;
      }
#pragma unroll
      for (int off = 8; off >= 1; off >>= 1)
        ps += __shfl_xor(ps, off, 32);
      l_i[r] = l_i[r] * corr + ps;
#pragma unroll
      for (int t = 0; t < 4; ++t) acc_o[t][r] *= corr;
    }

    // ---- restage P through LDS to re-fragment it as the A operand of P*V
    _Float16* pw = &Pl[wave * 16 * 72];
    __syncthreads();
#pragma unroll
    for (int t = 0; t < 4; ++t)
#pragma unroll
      for (int r = 0; r < 8; ++r)
        pw[(mh * 8 + r) * 72 + t * 16 + nlo] = (_Float16)pv[t][r];
    __syncthreads();

    // ---- O(16x64) += P(16x64) * V(64x64): 2 k-steps x 4 d-tiles
#pragma unroll
    for (int kk = 0; kk < 2; ++kk) {
      v16h ap = frag_a(pw + kk * 32, lane, 72);
#pragma unroll
      for (int t = 0; t < 4; ++t) {
        v16h bv = frag_b(&VtT[(t * 16) * 72 + kk * 32], lane, 72);
        acc_o[t] = __builtin_amdgcn_wmma_f32_16x16x32_f16(
            false, ap, false, bv, (short)0, acc_o[t], false, false);
      }
    }
  }

  // ---- normalize and write context [b, s, h*64 + d] (f16)
#pragma unroll
  for (int t = 0; t < 4; ++t) {
#pragma unroll
    for (int r = 0; r < 8; ++r) {
      int s = qbase + mh * 8 + r;
      float ov = acc_o[t][r] / l_i[r];
      ctx[((size_t)b * SEQ + s) * DIMN + h * HEAD_DIM + t * 16 + nlo] = (_Float16)ov;
    }
  }
}

// ---------------------------------------------------------------------------
// Launcher
// ---------------------------------------------------------------------------
extern "C" void kernel_launch(void* const* d_in, const int* in_sizes, int n_in,
                              void* d_out, int out_size, void* d_ws, size_t ws_size,
                              hipStream_t stream) {
  const float* q  = (const float*)d_in[0];
  const float* k  = (const float*)d_in[1];
  const float* v  = (const float*)d_in[2];
  const float* Wq = (const float*)d_in[3];
  const float* bq = (const float*)d_in[4];
  const float* Wk = (const float*)d_in[5];
  const float* bk = (const float*)d_in[6];
  const float* Wv = (const float*)d_in[7];
  const float* bv = (const float*)d_in[8];
  const float* Wo = (const float*)d_in[9];
  const float* bo = (const float*)d_in[10];
  const int* mask = (const int*)d_in[11];
  float* out = (float*)d_out;

  char* ws = (char*)d_ws;
  const size_t szA = (size_t)MTOT * DIMN * 2;  // 16 MB (f16 activation matrix)
  const size_t szW = (size_t)DIMN * DIMN * 2;  // 2 MB  (f16 weight matrix)
  _Float16* q16  = (_Float16*)(ws + 0 * szA);
  _Float16* k16  = (_Float16*)(ws + 1 * szA);
  _Float16* v16  = (_Float16*)(ws + 2 * szA);
  _Float16* Wq16 = (_Float16*)(ws + 3 * szA + 0 * szW);
  _Float16* Wk16 = (_Float16*)(ws + 3 * szA + 1 * szW);
  _Float16* Wv16 = (_Float16*)(ws + 3 * szA + 2 * szW);
  _Float16* Wo16 = (_Float16*)(ws + 3 * szA + 3 * szW);
  char* base2 = ws + 3 * szA + 4 * szW;
  _Float16* qh  = (_Float16*)(base2 + 0 * szA);   // [B,H,S,Dh]
  _Float16* kh  = (_Float16*)(base2 + 1 * szA);
  _Float16* vh  = (_Float16*)(base2 + 2 * szA);
  _Float16* ctx = (_Float16*)(base2 + 3 * szA);   // [B,S,DIM]
  unsigned* mb  = (unsigned*)(base2 + 4 * szA);   // packed mask bits (2 MB)

  // 1) f32 -> f16 conversions
  {
    int n4 = MTOT * DIMN / 4;
    int nb = (n4 + 255) / 256;
    cvt_f16_x4<<<nb, 256, 0, stream>>>((const float4*)q, q16, n4);
    cvt_f16_x4<<<nb, 256, 0, stream>>>((const float4*)k, k16, n4);
    cvt_f16_x4<<<nb, 256, 0, stream>>>((const float4*)v, v16, n4);
    int w4 = DIMN * DIMN / 4;
    int wb = (w4 + 255) / 256;
    cvt_f16_x4<<<wb, 256, 0, stream>>>((const float4*)Wq, Wq16, w4);
    cvt_f16_x4<<<wb, 256, 0, stream>>>((const float4*)Wk, Wk16, w4);
    cvt_f16_x4<<<wb, 256, 0, stream>>>((const float4*)Wv, Wv16, w4);
    cvt_f16_x4<<<wb, 256, 0, stream>>>((const float4*)Wo, Wo16, w4);
  }

  // 2) pack mask to bits
  {
    int nwords = BATCH * SEQ * (SEQ / 32);
    pack_mask<<<(nwords + 255) / 256, 256, 0, stream>>>(mask, mb, nwords);
  }

  // 3) QKV projections (WMMA GEMM, scatter to head layout)
  dim3 gg(DIMN / 128, MTOT / 128);
  gemm_wmma<1><<<gg, 256, 0, stream>>>(q16, Wq16, bq, (void*)qh, MTOT, DIMN, DIMN);
  gemm_wmma<1><<<gg, 256, 0, stream>>>(k16, Wk16, bk, (void*)kh, MTOT, DIMN, DIMN);
  gemm_wmma<1><<<gg, 256, 0, stream>>>(v16, Wv16, bv, (void*)vh, MTOT, DIMN, DIMN);

  // 4) attention
  attn_wmma<<<BATCH * HEADS * (SEQ / 128), 256, 0, stream>>>(qh, kh, vh, mb, ctx);

  // 5) output projection -> f32 d_out
  gemm_wmma<0><<<gg, 256, 0, stream>>>(ctx, Wo16, bo, (void*)out, MTOT, DIMN, DIMN);
}
